// Decoder_63196148793722
// MI455X (gfx1250) — matrix operand order repaired and checked by reading, not compiled, not run on hardware
//
#include <hip/hip_runtime.h>
#include <hip/hip_bf16.h>
#include <math.h>

// ---------------------------------------------------------------------------
// Tacotron2 decoder for MI455X (gfx1250, wave32, WMMA).
//
// The scan is weight-streaming bound (~37 MB/step in bf16 vs 570 MFLOP/step,
// weights L2-resident in the 192MB L2). All weights are converted once to
// bf16 and packed into WMMA-B fragment-major layout (32B contiguous per lane,
// loaded non-temporally: read once per step, never WGP$-resident). The
// sequential 256-step loop runs in ONE persistent 1024-thread workgroup with
// LDS-resident h0/h1/ctx/cum (bank-conflict-padded rows, ds_load_b128
// fragment loads) and register-resident LSTM cell state; gate columns are
// permuted so each wave owns i/f/g/o of its 32 cells. Matrix ops use
// v_wmma_f32_16x16x32_bf16 (M=16 == batch).
// ---------------------------------------------------------------------------

typedef __attribute__((ext_vector_type(16))) __bf16 v16bf;
typedef __attribute__((ext_vector_type(8)))  __bf16 v8bf;
typedef __attribute__((ext_vector_type(8)))  float  v8f;

#define B_      16
#define T_IN_   256
#define T_OUT_  256
#define N_MELS_ 80
#define ENC_    512
#define PRE_    256
#define DR_     1024
#define ATTN_   128
#define NF_     32
#define KS_     31

// LDS activation row strides: +8 bf16 (4 dwords) pad -> 4-bank rotation per
// row, so the 16 rows of an A-fragment hit distinct bank groups.
#define XS_S  (PRE_ + 8)   // 264
#define CTX_S (ENC_ + 8)   // 520
#define H_S   (DR_ + 8)    // 1032

__device__ __forceinline__ __bf16 f2bf(float f) {
  unsigned u = __builtin_bit_cast(unsigned, f);
  unsigned short h = (unsigned short)((u + 0x7FFFu + ((u >> 16) & 1u)) >> 16);
  return __builtin_bit_cast(__bf16, h);
}
__device__ __forceinline__ float bf2f(__bf16 b) {
  unsigned u = ((unsigned)__builtin_bit_cast(unsigned short, b)) << 16;
  return __builtin_bit_cast(float, u);
}
__device__ __forceinline__ float sigm(float x) { return 1.0f / (1.0f + __expf(-x)); }

__device__ __forceinline__ v8f wmma_bf16(v16bf a, v16bf b, v8f c) {
  return __builtin_amdgcn_wmma_f32_16x16x32_bf16(false, a, false, b, (short)0, c,
                                                 false, false);
}

// A-fragment (16x32 bf16, M x K) from row-major padded LDS.
// ISA layout: lane<16 holds K[0..7]+K[16..23] of row=lane; lane>=16 holds
// K[8..15]+K[24..31] of row=lane-16. Both halves are 16B contiguous and
// 16B aligned -> two ds_load_b128 per lane.
__device__ __forceinline__ v16bf load_a_frag(const __bf16* seg, int stride,
                                             int koff, int lane) {
  const int r  = lane & 15;
  const int kb = (lane >> 4) << 3;
  const __bf16* p = seg + r * stride + koff + kb;
  v8bf lo = *(const v8bf*)(p);
  v8bf hi = *(const v8bf*)(p + 16);
  return __builtin_shufflevector(lo, hi, 0, 1, 2, 3, 4, 5, 6, 7,
                                 8, 9, 10, 11, 12, 13, 14, 15);
}

// B-fragment from fragment-major packed weights: one 32B contiguous chunk per
// lane, non-temporal (streamed once per step; bypass WGP$, stay in L2).
__device__ __forceinline__ v16bf load_b_frag(const __bf16* W, int kt, int nt,
                                             int NT, int lane) {
  return __builtin_nontemporal_load(
      (const v16bf*)(W + ((((size_t)kt * NT + nt) * 32 + lane) << 4)));
}

// Gate-column permutation: dst col d (0..4095) in wave-chunked i/f/g/o order
// -> src col in PyTorch [i|f|g|o] order.
__device__ __forceinline__ int gate_perm_src(int d) {
  int chunk = d >> 7, local = d & 127;
  int g = local >> 5, m = local & 31;
  return (g << 10) + (chunk << 5) + m;
}

// ---------------- weight packing kernels (run once, parallel) --------------

__global__ void k_pack_gatesW(__bf16* __restrict__ dst,
                              const float* __restrict__ Wih,
                              const float* __restrict__ Whh,
                              int rows_ih, int KT, long total) {
  long idx = (long)blockIdx.x * blockDim.x + threadIdx.x;
  if (idx >= total) return;
  int e    = (int)(idx & 15);
  int lane = (int)((idx >> 4) & 31);
  long fi  = idx >> 9;
  int nt   = (int)(fi % 256);
  int kt   = (int)(fi / 256);
  int k    = kt * 32 + ((lane >> 4) << 3) + (e & 7) + ((e >> 3) << 4);
  int nd   = nt * 16 + (lane & 15);
  int ns   = gate_perm_src(nd);
  float v  = (k < rows_ih) ? Wih[(size_t)k * 4096 + ns]
                           : Whh[(size_t)(k - rows_ih) * 4096 + ns];
  dst[idx] = f2bf(v);
}

__global__ void k_pack_plainW(__bf16* __restrict__ dst,
                              const float* __restrict__ W,
                              int KT, int NT, int N, long total) {
  long idx = (long)blockIdx.x * blockDim.x + threadIdx.x;
  if (idx >= total) return;
  int e    = (int)(idx & 15);
  int lane = (int)((idx >> 4) & 31);
  long fi  = idx >> 9;
  int nt   = (int)(fi % NT);
  int kt   = (int)(fi / NT);
  int k    = kt * 32 + ((lane >> 4) << 3) + (e & 7) + ((e >> 3) << 4);
  int n    = nt * 16 + (lane & 15);
  dst[idx] = f2bf(W[(size_t)k * N + n]);
}

// Misc: permuted biases, fused conv kernel CL[k,a]=sum_f Cw[f,k]*Ld[f,a],
// CLb[a]=sum_f Cb[f]*Ld[f,a], bf16 copy of Wg.
__global__ void k_misc(float* b0p, float* b1p, float* CL, float* CLb,
                       __bf16* WgB, const float* b0, const float* b1,
                       const float* Cw, const float* Cb, const float* Ld,
                       const float* Wg) {
  int idx = blockIdx.x * 256 + threadIdx.x;
  if (idx < 4096) { b0p[idx] = b0[gate_perm_src(idx)]; return; }
  idx -= 4096;
  if (idx < 4096) { b1p[idx] = b1[gate_perm_src(idx)]; return; }
  idx -= 4096;
  if (idx < KS_ * ATTN_) {
    int k = idx >> 7, a = idx & 127;
    float s = 0.f;
    for (int f = 0; f < NF_; ++f) s += Cw[f * KS_ + k] * Ld[f * ATTN_ + a];
    CL[idx] = s;
    return;
  }
  idx -= KS_ * ATTN_;
  if (idx < ATTN_) {
    float s = 0.f;
    for (int f = 0; f < NF_; ++f) s += Cb[f] * Ld[f * ATTN_ + idx];
    CLb[idx] = s;
    return;
  }
  idx -= ATTN_;
  if (idx < DR_ + ENC_) WgB[idx] = f2bf(Wg[idx]);
}

// ---------------- prenet: all 256 steps in parallel ------------------------
__global__ __launch_bounds__(256) void k_prenet(
    __bf16* __restrict__ preB, const float* __restrict__ targets,
    const float* __restrict__ Wp1, const float* __restrict__ bp1,
    const float* __restrict__ Wp2, const float* __restrict__ bp2,
    const float* __restrict__ drop1, const float* __restrict__ drop2) {
  __shared__ float xin[N_MELS_];
  __shared__ float x1[PRE_];
  const int t = blockIdx.x >> 4, b = blockIdx.x & 15, j = threadIdx.x;
  if (j < N_MELS_)
    xin[j] = (t == 0) ? 0.f
                      : targets[(size_t)b * N_MELS_ * T_OUT_ + j * T_OUT_ + (t - 1)];
  __syncthreads();
  float s = bp1[j];
  for (int m = 0; m < N_MELS_; ++m) s += xin[m] * Wp1[m * PRE_ + j];
  s = fmaxf(s, 0.f) * drop1[(size_t)blockIdx.x * PRE_ + j];
  x1[j] = s;
  __syncthreads();
  float s2 = bp2[j];
  for (int k = 0; k < PRE_; ++k) s2 += x1[k] * Wp2[k * PRE_ + j];
  s2 = fmaxf(s2, 0.f) * drop2[(size_t)blockIdx.x * PRE_ + j];
  preB[(size_t)blockIdx.x * PRE_ + j] = f2bf(s2);
}

// ---------------- pmem = memory @ Wm + bm (WMMA) ---------------------------
__global__ __launch_bounds__(256) void k_pmem(float* __restrict__ pmem,
                                              const float* __restrict__ mem,
                                              const __bf16* __restrict__ WmF,
                                              const float* __restrict__ bm) {
  const int lane = threadIdx.x & 31;
  const int nt   = threadIdx.x >> 5;  // 8 waves -> 8 n-tiles
  const int mt   = blockIdx.x;        // 256 m-tiles of 16 rows (B*T_IN=4096)
  v8f acc = {0, 0, 0, 0, 0, 0, 0, 0};
  const int r = lane & 15, kb = (lane >> 4) << 3;
#pragma unroll 2
  for (int kt = 0; kt < ENC_ / 32; ++kt) {
    const float* p = mem + (size_t)(mt * 16 + r) * ENC_ + kt * 32;
    v16bf a;
#pragma unroll
    for (int i = 0; i < 8; ++i) a[i] = f2bf(p[kb + i]);
#pragma unroll
    for (int i = 0; i < 8; ++i) a[8 + i] = f2bf(p[kb + 16 + i]);
    v16bf b = load_b_frag(WmF, kt, nt, 8, lane);
    acc = wmma_bf16(a, b, acc);
  }
  const int col = nt * 16 + (lane & 15);
  const int rb  = (lane >> 4) << 3;
#pragma unroll
  for (int v = 0; v < 8; ++v)
    pmem[(size_t)(mt * 16 + rb + v) * ATTN_ + col] = acc[v] + bm[col];
}

// ---------------- persistent sequential decoder ----------------------------
__global__ __launch_bounds__(1024) void k_decoder(
    const __bf16* __restrict__ preB, const __bf16* __restrict__ W0f,
    const float* __restrict__ b0p, const __bf16* __restrict__ W1f,
    const float* __restrict__ b1p, const __bf16* __restrict__ WqF,
    const float* __restrict__ bq, const float* __restrict__ pmem,
    const float* __restrict__ CLg, const float* __restrict__ CLbg,
    const float* __restrict__ Vw, const float* __restrict__ Vb,
    const int* __restrict__ memlen, const float* __restrict__ mem,
    const __bf16* __restrict__ WpF, const float* __restrict__ bproj,
    const __bf16* __restrict__ WgB, const float* __restrict__ bg,
    float* __restrict__ melO, float* __restrict__ gateO,
    float* __restrict__ alignO) {
  extern __shared__ __align__(16) char smem_raw[];
  __bf16* xs   = (__bf16*)smem_raw;      // 16 x XS_S  (pad-strided)
  __bf16* ctxs = xs + B_ * XS_S;         // 16 x CTX_S
  __bf16* h0s  = ctxs + B_ * CTX_S;      // 16 x H_S
  __bf16* h1s  = h0s + B_ * H_S;         // 16 x H_S
  float* cum  = (float*)(h1s + B_ * H_S);  // 16 x 256
  float* aw   = cum + B_ * T_IN_;          // 16 x 256
  float* qs   = aw + B_ * T_IN_;           // 16 x 128
  float* es   = qs + B_ * ATTN_;           // 16 x 256
  float* CLs  = es + B_ * T_IN_;           // 31 x 128
  float* CLbs = CLs + KS_ * ATTN_;         // 128

  const int tid = threadIdx.x;
  const int lane = tid & 31;
  const int wave = tid >> 5;  // 32 waves
  const int col16 = lane & 15;
  const int rbase = (lane >> 4) << 3;

  // ---- init state (zero padded rows too) ----
  for (int i = tid; i < B_ * CTX_S; i += 1024) ctxs[i] = f2bf(0.f);
  for (int i = tid; i < B_ * H_S; i += 1024) { h0s[i] = f2bf(0.f); h1s[i] = f2bf(0.f); }
  for (int i = tid; i < B_ * XS_S; i += 1024) xs[i] = f2bf(0.f);
  for (int i = tid; i < B_ * T_IN_; i += 1024) cum[i] = 0.f;
  for (int i = tid; i < KS_ * ATTN_; i += 1024) CLs[i] = CLg[i];
  if (tid < ATTN_) CLbs[tid] = CLbg[tid];
  float c0r[2][8], c1r[2][8];
#pragma unroll
  for (int p = 0; p < 2; ++p)
#pragma unroll
    for (int v = 0; v < 8; ++v) { c0r[p][v] = 0.f; c1r[p][v] = 0.f; }
  __syncthreads();

  for (int t = 0; t < T_OUT_; ++t) {
    // stage x_t into LDS; prefetch next step's prenet row (global_prefetch_b8)
    for (int i = tid; i < B_ * PRE_; i += 1024)
      xs[(i >> 8) * XS_S + (i & 255)] = preB[(size_t)t * B_ * PRE_ + i];
    if (tid == 0) {
      int tn = (t + 1 < T_OUT_) ? t + 1 : t;
      __builtin_prefetch(preB + (size_t)tn * B_ * PRE_, 0, 1);
    }
    __syncthreads();

    // ============== LSTM cell 0: gates = [x | ctx | h0] @ W0cat ==========
    v8f acc[8];
#pragma unroll
    for (int nt = 0; nt < 8; ++nt) { v8f z = {0,0,0,0,0,0,0,0}; acc[nt] = z; }
#pragma unroll 2
    for (int kt = 0; kt < 56; ++kt) {
      const __bf16* seg; int stride, koff;
      if (kt < 8)       { seg = xs;   stride = XS_S;  koff = kt * 32; }
      else if (kt < 24) { seg = ctxs; stride = CTX_S; koff = (kt - 8) * 32; }
      else              { seg = h0s;  stride = H_S;   koff = (kt - 24) * 32; }
      v16bf a = load_a_frag(seg, stride, koff, lane);
#pragma unroll
      for (int nt = 0; nt < 8; ++nt) {
        v16bf b = load_b_frag(W0f, kt, wave * 8 + nt, 256, lane);
        acc[nt] = wmma_bf16(a, b, acc[nt]);
      }
    }
    __syncthreads();  // all reads of old h0s done before overwrite
#pragma unroll
    for (int p = 0; p < 2; ++p) {
      const int localc = p * 16 + col16;  // cell index within wave's 32
      const float bi = b0p[wave * 128 + 0 * 32 + localc];
      const float bf = b0p[wave * 128 + 1 * 32 + localc];
      const float bG = b0p[wave * 128 + 2 * 32 + localc];
      const float bo = b0p[wave * 128 + 3 * 32 + localc];
#pragma unroll
      for (int v = 0; v < 8; ++v) {
        float gi = acc[0 + p][v] + bi;
        float gf = acc[2 + p][v] + bf;
        float gg = acc[4 + p][v] + bG;
        float go = acc[6 + p][v] + bo;
        float c = sigm(gf) * c0r[p][v] + sigm(gi) * tanhf(gg);
        c0r[p][v] = c;
        h0s[(rbase + v) * H_S + wave * 32 + localc] = f2bf(sigm(go) * tanhf(c));
      }
    }
    __syncthreads();

    // ============== LSTM cell 1: gates = [h0 | ctx | h1] @ W1cat =========
#pragma unroll
    for (int nt = 0; nt < 8; ++nt) { v8f z = {0,0,0,0,0,0,0,0}; acc[nt] = z; }
#pragma unroll 2
    for (int kt = 0; kt < 80; ++kt) {
      const __bf16* seg; int stride, koff;
      if (kt < 32)      { seg = h0s;  stride = H_S;   koff = kt * 32; }
      else if (kt < 48) { seg = ctxs; stride = CTX_S; koff = (kt - 32) * 32; }
      else              { seg = h1s;  stride = H_S;   koff = (kt - 48) * 32; }
      v16bf a = load_a_frag(seg, stride, koff, lane);
#pragma unroll
      for (int nt = 0; nt < 8; ++nt) {
        v16bf b = load_b_frag(W1f, kt, wave * 8 + nt, 256, lane);
        acc[nt] = wmma_bf16(a, b, acc[nt]);
      }
    }
    __syncthreads();
#pragma unroll
    for (int p = 0; p < 2; ++p) {
      const int localc = p * 16 + col16;
      const float bi = b1p[wave * 128 + 0 * 32 + localc];
      const float bf = b1p[wave * 128 + 1 * 32 + localc];
      const float bG = b1p[wave * 128 + 2 * 32 + localc];
      const float bo = b1p[wave * 128 + 3 * 32 + localc];
#pragma unroll
      for (int v = 0; v < 8; ++v) {
        float gi = acc[0 + p][v] + bi;
        float gf = acc[2 + p][v] + bf;
        float gg = acc[4 + p][v] + bG;
        float go = acc[6 + p][v] + bo;
        float c = sigm(gf) * c1r[p][v] + sigm(gi) * tanhf(gg);
        c1r[p][v] = c;
        h1s[(rbase + v) * H_S + wave * 32 + localc] = f2bf(sigm(go) * tanhf(c));
      }
    }
    __syncthreads();

    // ============== attention query: q = h1 @ Wq + bq (waves 0..7) =======
    if (wave < 8) {
      v8f qa = {0,0,0,0,0,0,0,0};
#pragma unroll 2
      for (int kt = 0; kt < 32; ++kt) {
        v16bf a = load_a_frag(h1s, H_S, kt * 32, lane);
        v16bf b = load_b_frag(WqF, kt, wave, 8, lane);
        qa = wmma_bf16(a, b, qa);
      }
#pragma unroll
      for (int v = 0; v < 8; ++v)
        qs[(rbase + v) * ATTN_ + wave * 16 + col16] = qa[v] + bq[wave * 16 + col16];
    }
    __syncthreads();

    // ============== energies e[b,i] (fused conv via CL) ==================
    {
      const float vb = Vb[0];
      for (int r = 0; r < 4; ++r) {
        int pi = r * 1024 + tid;  // 0..4095
        int b = pi >> 8, i = pi & 255;
        float w[KS_];
#pragma unroll
        for (int k = 0; k < KS_; ++k) {
          int j = i + k - 15;
          w[k] = (j >= 0 && j < T_IN_) ? cum[b * T_IN_ + j] : 0.f;
        }
        float e = vb;
        const float* prow = pmem + (size_t)(b * T_IN_ + i) * ATTN_;
        const float* qrow = qs + b * ATTN_;
        for (int a = 0; a < ATTN_; ++a) {
          float loc = CLbs[a];
#pragma unroll
          for (int k = 0; k < KS_; ++k) loc += w[k] * CLs[k * ATTN_ + a];
          e += tanhf(qrow[a] + prow[a] + loc) * Vw[a];
        }
        if (i >= memlen[b]) e = -1e30f;
        es[pi] = e;
      }
    }
    __syncthreads();

    // ============== masked softmax per batch row (waves 0..15) ===========
    if (wave < 16) {
      const int b = wave;
      float ev[8], vmax = -1e30f;
#pragma unroll
      for (int j = 0; j < 8; ++j) {
        ev[j] = es[b * T_IN_ + j * 32 + lane];
        vmax = fmaxf(vmax, ev[j]);
      }
#pragma unroll
      for (int off = 16; off; off >>= 1) vmax = fmaxf(vmax, __shfl_xor(vmax, off, 32));
      float s = 0.f;
#pragma unroll
      for (int j = 0; j < 8; ++j) { ev[j] = __expf(ev[j] - vmax); s += ev[j]; }
#pragma unroll
      for (int off = 16; off; off >>= 1) s += __shfl_xor(s, off, 32);
      float inv = 1.f / s;
#pragma unroll
      for (int j = 0; j < 8; ++j) {
        int i = j * 32 + lane;
        float a = ev[j] * inv;
        aw[b * T_IN_ + i] = a;
        cum[b * T_IN_ + i] += a;
        alignO[(size_t)b * T_IN_ * T_OUT_ + (size_t)i * T_OUT_ + t] = a;
      }
    }
    __syncthreads();

    // ============== ctx[b,e] = sum_i aw[b,i] * memory[b,i,e] =============
    for (int r = 0; r < 8; ++r) {
      int oi = r * 1024 + tid;  // 0..8191
      int b = oi >> 9, e = oi & 511;
      const float* ar = aw + b * T_IN_;
      const float* mr = mem + (size_t)b * T_IN_ * ENC_ + e;
      float s = 0.f;
      for (int i = 0; i < T_IN_; ++i) s += ar[i] * mr[(size_t)i * ENC_];
      ctxs[b * CTX_S + e] = f2bf(s);
    }
    __syncthreads();

    // ============== outputs: mel = [h1|ctx]@Wproj, gate = z.Wg ===========
    if (wave < 5) {
      v8f ma = {0,0,0,0,0,0,0,0};
#pragma unroll 2
      for (int kt = 0; kt < 48; ++kt) {
        const __bf16* seg; int stride, koff;
        if (kt < 32) { seg = h1s; stride = H_S; koff = kt * 32; }
        else         { seg = ctxs; stride = CTX_S; koff = (kt - 32) * 32; }
        v16bf a = load_a_frag(seg, stride, koff, lane);
        v16bf b = load_b_frag(WpF, kt, wave, 5, lane);
        ma = wmma_bf16(a, b, ma);
      }
      const int m = wave * 16 + col16;  // mel channel (< 80)
#pragma unroll
      for (int v = 0; v < 8; ++v)
        melO[(size_t)(rbase + v) * N_MELS_ * T_OUT_ + (size_t)m * T_OUT_ + t] =
            ma[v] + bproj[m];
    } else if (wave == 5 && lane < B_) {
      float g = bg[0];
      for (int k = 0; k < DR_; ++k)
        g += bf2f(h1s[lane * H_S + k]) * bf2f(WgB[k]);
      for (int k = 0; k < ENC_; ++k)
        g += bf2f(ctxs[lane * CTX_S + k]) * bf2f(WgB[DR_ + k]);
      gateO[lane * T_OUT_ + t] = g;
    }
    __syncthreads();
  }
}

// ---------------------------------------------------------------------------
extern "C" void kernel_launch(void* const* d_in, const int* in_sizes, int n_in,
                              void* d_out, int out_size, void* d_ws,
                              size_t ws_size, hipStream_t stream) {
  (void)in_sizes; (void)n_in; (void)out_size; (void)ws_size;
  const float* mem     = (const float*)d_in[0];
  const float* targets = (const float*)d_in[1];
  const int*   memlen  = (const int*)d_in[2];
  const float* Wp1  = (const float*)d_in[3];
  const float* bp1  = (const float*)d_in[4];
  const float* Wp2  = (const float*)d_in[5];
  const float* bp2  = (const float*)d_in[6];
  const float* Wih0 = (const float*)d_in[7];
  const float* Whh0 = (const float*)d_in[8];
  const float* b0   = (const float*)d_in[9];
  const float* Wih1 = (const float*)d_in[10];
  const float* Whh1 = (const float*)d_in[11];
  const float* b1   = (const float*)d_in[12];
  const float* Wq   = (const float*)d_in[13];
  const float* bq   = (const float*)d_in[14];
  const float* Wm   = (const float*)d_in[15];
  const float* bm   = (const float*)d_in[16];
  const float* Vw   = (const float*)d_in[17];
  const float* Vb   = (const float*)d_in[18];
  const float* Cw   = (const float*)d_in[19];
  const float* Cb   = (const float*)d_in[20];
  const float* Ld   = (const float*)d_in[21];
  const float* Wproj= (const float*)d_in[22];
  const float* bproj= (const float*)d_in[23];
  const float* Wg   = (const float*)d_in[24];
  const float* bg   = (const float*)d_in[25];
  const float* drop1= (const float*)d_in[26];
  const float* drop2= (const float*)d_in[27];

  // workspace carve-up (~40 MB), 256B-aligned chunks
  char* ws = (char*)d_ws;
  size_t off = 0;
  auto take = [&](size_t bytes) -> char* {
    char* p = ws + off;
    off += (bytes + 255) & ~(size_t)255;
    return p;
  };
  __bf16* W0f  = (__bf16*)take((size_t)56 * 256 * 512 * 2);
  __bf16* W1f  = (__bf16*)take((size_t)80 * 256 * 512 * 2);
  __bf16* WqF  = (__bf16*)take((size_t)32 * 8 * 512 * 2);
  __bf16* WpF  = (__bf16*)take((size_t)48 * 5 * 512 * 2);
  __bf16* WmF  = (__bf16*)take((size_t)16 * 8 * 512 * 2);
  __bf16* WgB  = (__bf16*)take((size_t)(DR_ + ENC_) * 2);
  float*  b0p  = (float*)take(4096 * 4);
  float*  b1p  = (float*)take(4096 * 4);
  float*  CLp  = (float*)take(KS_ * ATTN_ * 4);
  float*  CLbp = (float*)take(ATTN_ * 4);
  __bf16* preB = (__bf16*)take((size_t)T_OUT_ * B_ * PRE_ * 2);
  float*  pmemP= (float*)take((size_t)B_ * T_IN_ * ATTN_ * 4);

  float* melO   = (float*)d_out;                           // (B, N_MELS, T_OUT)
  float* gateO  = melO + (size_t)B_ * N_MELS_ * T_OUT_;    // (B, T_OUT)
  float* alignO = gateO + (size_t)B_ * T_OUT_;             // (B, T_IN, T_OUT)

  // --- parallel packing / precompute ---
  {
    long tot0 = (long)56 * 256 * 512;
    k_pack_gatesW<<<(int)((tot0 + 255) / 256), 256, 0, stream>>>(W0f, Wih0, Whh0,
                                                                 768, 56, tot0);
    long tot1 = (long)80 * 256 * 512;
    k_pack_gatesW<<<(int)((tot1 + 255) / 256), 256, 0, stream>>>(W1f, Wih1, Whh1,
                                                                 1536, 80, tot1);
    long tq = (long)32 * 8 * 512;
    k_pack_plainW<<<(int)((tq + 255) / 256), 256, 0, stream>>>(WqF, Wq, 32, 8, 128, tq);
    long tp = (long)48 * 5 * 512;
    k_pack_plainW<<<(int)((tp + 255) / 256), 256, 0, stream>>>(WpF, Wproj, 48, 5, 80, tp);
    long tm = (long)16 * 8 * 512;
    k_pack_plainW<<<(int)((tm + 255) / 256), 256, 0, stream>>>(WmF, Wm, 16, 8, 128, tm);
    int misc_total = 4096 + 4096 + KS_ * ATTN_ + ATTN_ + (DR_ + ENC_);
    k_misc<<<(misc_total + 255) / 256, 256, 0, stream>>>(b0p, b1p, CLp, CLbp, WgB,
                                                         b0, b1, Cw, Cb, Ld, Wg);
    k_prenet<<<T_OUT_ * B_, 256, 0, stream>>>(preB, targets, Wp1, bp1, Wp2, bp2,
                                              drop1, drop2);
    k_pmem<<<256, 256, 0, stream>>>(pmemP, mem, WmF, bm);
  }

  // --- persistent sequential decoder: 1 WGP, 32 waves, ~161KB LDS ---
  const size_t smem =
      (size_t)(B_ * XS_S + B_ * CTX_S + 2 * B_ * H_S) * 2 +                 // bf16
      (size_t)(2 * B_ * T_IN_ + B_ * ATTN_ + B_ * T_IN_ + KS_ * ATTN_ + ATTN_) * 4;
  k_decoder<<<1, 1024, smem, stream>>>(preB, W0f, b0p, W1f, b1p, WqF, bq, pmemP,
                                       CLp, CLbp, Vw, Vb, memlen, mem, WpF,
                                       bproj, WgB, bg, melO, gateO, alignO);
}
